// GatedLinearRecurrence_57019985822493
// MI455X (gfx1250) — 16-bit and 8-bit matrix operand order repaired and checked by reading, compiled, not run on hardware
//
#include <hip/hip_runtime.h>
#include <stdint.h>
#include <stddef.h>

#define B_  8
#define S_  2048
#define H_  1024
#define BS_ (B_ * S_)

#define SW_ 64.0f
#define SH_ 16.0f

typedef __attribute__((ext_vector_type(8)))  float  v8f;
typedef __attribute__((ext_vector_type(16))) __bf16 v16bf;
typedef __attribute__((ext_vector_type(16))) int    v16i;

union FragBF { v16bf v; uint4 q[2]; };
union FragF8 { v16i  v; uint4 q[4]; uint2 d[8]; };
union AccF   { v8f   v; float f[8]; };
union Pack4  { uint2 u2; unsigned short s[4]; };

__device__ __forceinline__ unsigned short f2bfu(float f) {
  unsigned u = __float_as_uint(f);
  return (unsigned short)((u + 0x7FFFu + ((u >> 16) & 1u)) >> 16);
}

__device__ __forceinline__ unsigned char f2fp8(float f) {
  unsigned u = __float_as_uint(f);
  unsigned sign = (u >> 24) & 0x80u;
  float a = fabsf(f);
  if (!(a >= 0.001953125f)) return (unsigned char)sign;
  if (a >= 448.0f) return (unsigned char)(sign | 0x7Eu);
  int e; float m = frexpf(a, &e);
  int ef = e - 1 + 7;
  if (ef <= 0) {
    int mant = (int)(a * 512.0f + 0.5f);
    return (unsigned char)(sign | (mant & 7));
  }
  int mant = (int)(m * 16.0f + 0.5f) - 8;
  if (mant >= 8) { mant = 0; ef++; }
  if (ef >= 16) return (unsigned char)(sign | 0x7Eu);
  return (unsigned char)(sign | (ef << 3) | mant);
}

__device__ __forceinline__ float sigm_(float x) { return 1.0f / (1.0f + __expf(-x)); }
__device__ __forceinline__ float silu_(float x) { return x * sigm_(x); }

__global__ void __launch_bounds__(256) init_kernel(const float* __restrict__ h0,
                                                   float* __restrict__ hbuf,
                                                   unsigned* __restrict__ bar) {
  int idx = blockIdx.x * 256 + threadIdx.x;
  if (idx < 16 * H_) {
    int r = idx >> 10, c = idx & (H_ - 1);
    hbuf[idx]            = (r < B_) ? h0[r * H_ + c] : 0.0f;
    hbuf[16 * H_ + idx]  = 0.0f;
  }
  if (idx == 0) { bar[0] = 0u; bar[1] = 0u; }
}

__global__ void __launch_bounds__(256) quantw_kernel(const float* __restrict__ Wg,
                                                     const float* __restrict__ Wr,
                                                     unsigned* __restrict__ wq8) {
  int id = blockIdx.x * 256 + threadIdx.x;
  if (id >= 16 * 12 * 8 * 32 * 16) return;
  int j    = id & 15;
  int lane = (id >> 4) & 31;
  int kc   = (id >> 9) & 7;
  int t2   = id >> 12;
  int tile = t2 % 12;
  int wg   = t2 / 12;
  int half = lane >> 4, col16 = lane & 15;
  int kbase = kc * 128 + 32 * (j >> 2) + 16 * half + 4 * (j & 3);
  int grp = tile >> 2, ct = tile & 3;
  int col = wg * 64 + ct * 16 + col16;
  unsigned out = 0;
  #pragma unroll
  for (int b = 0; b < 4; ++b) {
    int k = kbase + b;
    float w;
    if (grp == 0)      w = Wg[(size_t)k * 2048 + col];
    else if (grp == 1) w = Wg[(size_t)k * 2048 + 1024 + col];
    else               w = Wr[(size_t)k * 1024 + col];
    out |= ((unsigned)f2fp8(w * SW_)) << (8 * b);
  }
  wq8[id] = out;
}

__global__ void __launch_bounds__(256, 1)
proj_in_kernel(const float* __restrict__ x, const float* __restrict__ W_in,
               const float* __restrict__ b_in, float* __restrict__ drive) {
  extern __shared__ char smem[];
  unsigned short* As = (unsigned short*)smem;
  unsigned short* Bs = (unsigned short*)(smem + 8192);
  float*          Cs = (float*)(smem + 8192 + 12288);

  const int tid  = threadIdx.x;
  const int wave = tid >> 5, lane = tid & 31;
  const int half = lane >> 4, l16 = lane & 15;
  const int mpart = wave & 3, npart = wave >> 2;
  const size_t mBase = (size_t)blockIdx.x * 128;
  const int dBase = blockIdx.y * 64;

  AccF acc[2][6];
  #pragma unroll
  for (int mt = 0; mt < 2; ++mt)
    #pragma unroll
    for (int nt = 0; nt < 6; ++nt)
      #pragma unroll
      for (int e = 0; e < 8; ++e) acc[mt][nt].f[e] = 0.0f;

  for (int kb = 0; kb < H_; kb += 32) {
    __syncthreads();
    for (int l4 = tid; l4 < 1024; l4 += 256) {
      int r = l4 >> 3, k4 = (l4 & 7) * 4;
      float4 xv = *(const float4*)&x[(mBase + r) * H_ + kb + k4];
      Pack4 p;
      p.s[0] = f2bfu(xv.x); p.s[1] = f2bfu(xv.y);
      p.s[2] = f2bfu(xv.z); p.s[3] = f2bfu(xv.w);
      *(uint2*)&As[r * 32 + k4] = p.u2;
    }
    for (int l4 = tid; l4 < 1536; l4 += 256) {
      int fq = l4 % 48, k = l4 / 48;
      int f = fq * 4;
      int grp = f >> 6, cc = f & 63;
      float4 wv = *(const float4*)&W_in[(size_t)(kb + k) * 3072 + grp * 1024 + dBase + cc];
      Bs[(f + 0) * 32 + k] = f2bfu(wv.x);
      Bs[(f + 1) * 32 + k] = f2bfu(wv.y);
      Bs[(f + 2) * 32 + k] = f2bfu(wv.z);
      Bs[(f + 3) * 32 + k] = f2bfu(wv.w);
    }
    __syncthreads();

    FragBF af[2];
    #pragma unroll
    for (int mt = 0; mt < 2; ++mt) {
      int r = mpart * 32 + mt * 16 + l16;
      af[mt].q[0] = *(const uint4*)&As[r * 32 + half * 8];
      af[mt].q[1] = *(const uint4*)&As[r * 32 + 16 + half * 8];
    }
    #pragma unroll
    for (int nt = 0; nt < 6; ++nt) {
      int f = npart * 96 + nt * 16 + l16;
      const uint4* p = (const uint4*)&Bs[f * 32 + half * 8];
      FragBF bf;
      bf.q[0] = p[0]; bf.q[1] = p[2];
      acc[0][nt].v = __builtin_amdgcn_wmma_f32_16x16x32_bf16(
          false, af[0].v, false, bf.v, (short)0, acc[0][nt].v, false, false);
      acc[1][nt].v = __builtin_amdgcn_wmma_f32_16x16x32_bf16(
          false, af[1].v, false, bf.v, (short)0, acc[1][nt].v, false, false);
    }
  }

  #pragma unroll
  for (int mt = 0; mt < 2; ++mt)
    #pragma unroll
    for (int nt = 0; nt < 6; ++nt)
      #pragma unroll
      for (int j = 0; j < 8; ++j) {
        int r = mpart * 32 + mt * 16 + j + 8 * half;
        int f = npart * 96 + nt * 16 + l16;
        Cs[r * 192 + f] = acc[mt][nt].f[j];
      }
  __syncthreads();

  for (int l = tid; l < 8192; l += 256) {
    int r = l >> 6, c = l & 63;
    int gc = dBase + c;
    float u = Cs[r * 192 + c]        + b_in[gc];
    float v = Cs[r * 192 + 64 + c]   + b_in[1024 + gc];
    float w = Cs[r * 192 + 128 + c]  + b_in[2048 + gc];
    drive[(mBase + r) * H_ + gc] = silu_(u) * silu_(v) + w;
  }
}

__global__ void __launch_bounds__(384, 1)
scan_kernel(const float* __restrict__ drive, const unsigned char* __restrict__ wq8,
            const float* __restrict__ b_gate, const float* __restrict__ b_rec,
            float* __restrict__ hbuf, unsigned short* __restrict__ hs,
            float* __restrict__ finalh, unsigned* __restrict__ bar) {
  extern __shared__ char smem[];
  unsigned char* Wl = (unsigned char*)smem;
  unsigned char* H8 = (unsigned char*)(smem + 196608);
  float*         Cs = (float*)(smem + 196608 + 16384);

  const int tid  = threadIdx.x;
  const int wave = tid >> 5;
  const int lane = tid & 31;
  const int half = lane >> 4, l16 = lane & 15;
  const int wg = blockIdx.x;
  const int colBase = wg * 64;
  const float invScale = 1.0f / (SW_ * SH_);

  {
    const unsigned char* gbase = wq8 + (size_t)wg * 196608;
    for (int i = tid; i < 196608 / 16; i += 384) {
      unsigned gofs = (unsigned)i * 16u;
      unsigned lofs = (unsigned)(unsigned long long)(Wl + gofs);
      asm volatile("global_load_async_to_lds_b128 %0, %1, %2"
                   :: "v"(lofs), "v"(gofs), "s"(gbase) : "memory");
    }
    asm volatile("s_wait_asynccnt 0" ::: "memory");
  }

  for (int t = 0; t < S_; ++t) {
    const float* hcur  = hbuf + (size_t)(t & 1) * 16 * H_;
    float*       hnext = hbuf + (size_t)((t + 1) & 1) * 16 * H_;

    __syncthreads();
    for (int l = tid; l < 4096; l += 384) {
      float4 hv = *(const float4*)&hcur[l * 4];
      unsigned o =  (unsigned)f2fp8(hv.x * SH_)
                 | ((unsigned)f2fp8(hv.y * SH_) << 8)
                 | ((unsigned)f2fp8(hv.z * SH_) << 16)
                 | ((unsigned)f2fp8(hv.w * SH_) << 24);
      ((unsigned*)H8)[l] = o;
    }
    __syncthreads();

    AccF acc;
    #pragma unroll
    for (int e = 0; e < 8; ++e) acc.f[e] = 0.0f;
    #pragma unroll
    for (int kc = 0; kc < 8; ++kc) {
      FragF8 a, b;
      #pragma unroll
      for (int blk = 0; blk < 2; ++blk)
        #pragma unroll
        for (int g = 0; g < 4; ++g) {
          int k = kc * 128 + blk * 64 + (g >> 1) * 32 + half * 16 + (g & 1) * 8;
          a.d[blk * 4 + g] = *(const uint2*)&H8[l16 * 1024 + k];
        }
      const uint4* wp = (const uint4*)&Wl[(((size_t)wave * 8 + kc) * 32 + lane) * 64];
      b.q[0] = wp[0]; b.q[1] = wp[1]; b.q[2] = wp[2]; b.q[3] = wp[3];
      acc.v = __builtin_amdgcn_wmma_f32_16x16x128_fp8_fp8(a.v, b.v, (short)0, acc.v, false, false);
    }
    #pragma unroll
    for (int j = 0; j < 8; ++j)
      Cs[wave * 256 + (j + 8 * half) * 16 + l16] = acc.f[j];
    __syncthreads();

    for (int l = tid; l < 512; l += 384) {
      int r = l >> 6, c = l & 63;
      int ct = c >> 4, cc = c & 15;
      int gc = colBase + c;
      float gv = Cs[(ct)     * 256 + r * 16 + cc] * invScale + b_gate[gc];
      float gt = Cs[(4 + ct) * 256 + r * 16 + cc] * invScale + b_gate[1024 + gc];
      float rc = Cs[(8 + ct) * 256 + r * 16 + cc] * invScale + b_rec[gc];
      float gate = sigm_(gv) * silu_(gt);
      size_t didx = ((size_t)r * S_ + t) * H_ + gc;
      float d = drive[didx];
      if (t + 1 < S_) __builtin_prefetch(&drive[didx + H_], 0, 0);
      float nh = gate * rc + (1.0f - gate) * d;
      hnext[r * H_ + gc] = nh;
      hs[didx] = f2bfu(nh);
      if (t == S_ - 1) finalh[r * H_ + gc] = nh;
    }

    __syncthreads();
    if (tid == 0) {
      __threadfence();
      unsigned g = __hip_atomic_load(&bar[1], __ATOMIC_ACQUIRE, __HIP_MEMORY_SCOPE_AGENT);
      unsigned a = __hip_atomic_fetch_add(&bar[0], 1u, __ATOMIC_ACQ_REL, __HIP_MEMORY_SCOPE_AGENT);
      if (a == 15u) {
        __hip_atomic_store(&bar[0], 0u, __ATOMIC_RELAXED, __HIP_MEMORY_SCOPE_AGENT);
        __hip_atomic_fetch_add(&bar[1], 1u, __ATOMIC_RELEASE, __HIP_MEMORY_SCOPE_AGENT);
      } else {
        while (__hip_atomic_load(&bar[1], __ATOMIC_ACQUIRE, __HIP_MEMORY_SCOPE_AGENT) == g)
          __builtin_amdgcn_s_sleep(1);
      }
    }
    if (tid < 32) __builtin_amdgcn_s_cluster_barrier();
    __syncthreads();
  }
}

__global__ void __launch_bounds__(256, 1)
proj_out_kernel(const unsigned short* __restrict__ hs, const float* __restrict__ W_out,
                const float* __restrict__ b_out, float* __restrict__ out) {
  __shared__ unsigned short As[128 * 32];
  __shared__ unsigned short Bs[128 * 32];

  const int tid  = threadIdx.x;
  const int wave = tid >> 5, lane = tid & 31;
  const int half = lane >> 4, l16 = lane & 15;
  const int mpart = wave & 1, npart = wave >> 1;
  const size_t mBase = (size_t)blockIdx.x * 128;
  const int nBase = blockIdx.y * 128;

  AccF acc[4][2];
  #pragma unroll
  for (int mt = 0; mt < 4; ++mt)
    #pragma unroll
    for (int nt = 0; nt < 2; ++nt)
      #pragma unroll
      for (int e = 0; e < 8; ++e) acc[mt][nt].f[e] = 0.0f;

  for (int kb = 0; kb < H_; kb += 32) {
    __syncthreads();
    for (int l8 = tid; l8 < 512; l8 += 256) {
      int r = l8 >> 2, k8 = (l8 & 3) * 8;
      *(uint4*)&As[r * 32 + k8] = *(const uint4*)&hs[(mBase + r) * H_ + kb + k8];
    }
    for (int l4 = tid; l4 < 1024; l4 += 256) {
      int cq = l4 & 31, k = l4 >> 5;
      int c = cq * 4;
      float4 wv = *(const float4*)&W_out[(size_t)(kb + k) * H_ + nBase + c];
      Bs[(c + 0) * 32 + k] = f2bfu(wv.x);
      Bs[(c + 1) * 32 + k] = f2bfu(wv.y);
      Bs[(c + 2) * 32 + k] = f2bfu(wv.z);
      Bs[(c + 3) * 32 + k] = f2bfu(wv.w);
    }
    __syncthreads();

    FragBF bf[2];
    #pragma unroll
    for (int nt = 0; nt < 2; ++nt) {
      int c = npart * 32 + nt * 16 + l16;
      const uint4* p = (const uint4*)&Bs[c * 32 + half * 8];
      bf[nt].q[0] = p[0]; bf[nt].q[1] = p[2];
    }
    #pragma unroll
    for (int mt = 0; mt < 4; ++mt) {
      int r = mpart * 64 + mt * 16 + l16;
      FragBF af;
      af.q[0] = *(const uint4*)&As[r * 32 + half * 8];
      af.q[1] = *(const uint4*)&As[r * 32 + 16 + half * 8];
      #pragma unroll
      for (int nt = 0; nt < 2; ++nt)
        acc[mt][nt].v = __builtin_amdgcn_wmma_f32_16x16x32_bf16(
            false, af.v, false, bf[nt].v, (short)0, acc[mt][nt].v, false, false);
    }
  }

  #pragma unroll
  for (int mt = 0; mt < 4; ++mt)
    #pragma unroll
    for (int nt = 0; nt < 2; ++nt)
      #pragma unroll
      for (int j = 0; j < 8; ++j) {
        size_t r = mBase + mpart * 64 + mt * 16 + j + 8 * half;
        int col = nBase + npart * 32 + nt * 16 + l16;
        out[r * H_ + col] = acc[mt][nt].f[j] + b_out[col];
      }
}

extern "C" void kernel_launch(void* const* d_in, const int* in_sizes, int n_in,
                              void* d_out, int out_size, void* d_ws, size_t ws_size,
                              hipStream_t stream) {
  (void)in_sizes; (void)n_in; (void)out_size; (void)ws_size;
  const float* x      = (const float*)d_in[0];
  const float* h0     = (const float*)d_in[1];
  const float* W_in   = (const float*)d_in[2];
  const float* b_in   = (const float*)d_in[3];
  const float* W_gate = (const float*)d_in[4];
  const float* b_gate = (const float*)d_in[5];
  const float* W_rec  = (const float*)d_in[6];
  const float* b_rec  = (const float*)d_in[7];
  const float* W_out  = (const float*)d_in[8];
  const float* b_out  = (const float*)d_in[9];

  float* out    = (float*)d_out;
  float* finalh = out + (size_t)BS_ * H_;

  char* ws = (char*)d_ws;
  float*          drive = (float*)ws;
  unsigned short* hsb   = (unsigned short*)(ws + ((size_t)64 << 20));
  unsigned char*  wq8   = (unsigned char*)(ws + ((size_t)96 << 20));
  float*          hbuf  = (float*)(ws + ((size_t)99 << 20));
  unsigned*       bar   = (unsigned*)(ws + ((size_t)99 << 20) + (128 << 10));

  init_kernel<<<64, 256, 0, stream>>>(h0, hbuf, bar);
  quantw_kernel<<<3072, 256, 0, stream>>>(W_gate, W_rec, (unsigned*)wq8);
  proj_in_kernel<<<dim3(128, 16), 256, 118784, stream>>>(x, W_in, b_in, drive);
  scan_kernel<<<16, 384, 225280, stream>>>(drive, wq8, b_gate, b_rec, hbuf, hsb, finalh, bar);
  proj_out_kernel<<<dim3(128, 8), 256, 0, stream>>>(hsb, W_out, b_out, out);
}
